// ResidualAttentionBlock_visual_55594056679801
// MI455X (gfx1250) — compile-verified, matrix operand order checked by reading, not run on hardware
//
#include <hip/hip_runtime.h>
#include <hip/hip_bf16.h>

// ---------------------------------------------------------------------------
// CLIP ResidualAttentionBlock forward for gfx1250 (MI455X, wave32, WMMA).
// All GEMMs run on v_wmma_f32_16x16x32_bf16 (bf16 A/B, f32 accumulate).
// GEMM tiles are staged to LDS with GLOBAL_LOAD_ASYNC_TO_LDS_B128 (ASYNCcnt
// path, builtin confirmed present on this toolchain), fallback plain b128.
// ---------------------------------------------------------------------------

typedef __attribute__((ext_vector_type(16))) __bf16 v16bf;
typedef __attribute__((ext_vector_type(8)))  __bf16 v8bf;
typedef __attribute__((ext_vector_type(8)))  float  v8f;

#define LDIM   257
#define NBATCH 64
#define DMODEL 768
#define NHEADS 12
#define HDIM   64
#define MROWS  (LDIM * NBATCH)   // 16448  (multiple of 64)
#define D3     (3 * DMODEL)      // 2304
#define LPAD   288               // 257 padded to a multiple of 32
#define NHPAIR (NBATCH * NHEADS) // 768

#if defined(__gfx1250__) && __has_builtin(__builtin_amdgcn_global_load_async_to_lds_b128)
#define HAVE_ASYNC_LDS 1
typedef int v4i __attribute__((ext_vector_type(4)));
typedef v4i __attribute__((address_space(1))) gv4i;   // global v4i
typedef v4i __attribute__((address_space(3))) lv4i;   // LDS v4i (32-bit ptr)
__device__ __forceinline__ void async_b128(const void* gsrc, void* ldst) {
  // as1 VA == generic VA (bit identical); as3 offset == low 32 bits of the
  // generic LDS address (flat aperture rule).
  __builtin_amdgcn_global_load_async_to_lds_b128(
      __builtin_bit_cast(gv4i*, gsrc),
      (lv4i*)(unsigned)(size_t)ldst, 0, 0);
}
__device__ __forceinline__ void wait_async0() {
#if __has_builtin(__builtin_amdgcn_s_wait_asynccnt)
  __builtin_amdgcn_s_wait_asynccnt(0);
#else
  asm volatile("s_wait_asynccnt 0x0" ::: "memory");
#endif
}
#else
#define HAVE_ASYNC_LDS 0
#endif

__device__ __forceinline__ __bf16 f2bf(float f) {
  unsigned u = __builtin_bit_cast(unsigned, f);
  u += 0x7FFFu + ((u >> 16) & 1u);            // round-to-nearest-even
  unsigned short h = (unsigned short)(u >> 16);
  return __builtin_bit_cast(__bf16, h);
}

// K index inside a 32-wide K-slice for element e of a 16-bf16 fragment,
// per ISA 7.12.2 (16-bit A 16x32): lanes 0-15 hold K {0..7,16..23},
// lanes 16-31 hold K {8..15,24..31}.  B uses the symmetric layout (lane = N).
// Elements 0..7 / 8..15 are each K-contiguous -> ds_load_b128 pairs.
__device__ __forceinline__ int kmap(int e, int hl) {
  return ((e >> 3) << 4) + (hl << 3) + (e & 7);
}

// ---------------------------------------------------------------------------
// fp32 -> bf16 weight conversion
// ---------------------------------------------------------------------------
__global__ __launch_bounds__(256) void cvt_bf16_kernel(const float* __restrict__ in,
                                                       __bf16* __restrict__ out, int n) {
  for (int i = blockIdx.x * 256 + threadIdx.x; i < n; i += gridDim.x * 256)
    out[i] = f2bf(in[i]);
}

// ---------------------------------------------------------------------------
// LayerNorm over D=768 per row; emits bf16 activations for the WMMA GEMMs.
// ---------------------------------------------------------------------------
__global__ __launch_bounds__(256) void ln_kernel(const float* __restrict__ x,
                                                 const float* __restrict__ g,
                                                 const float* __restrict__ b,
                                                 __bf16* __restrict__ out) {
  const int row = blockIdx.x;
  const int t   = threadIdx.x;
  __shared__ float s1[256], s2[256];
  float vals[3], v0 = 0.f, v1 = 0.f;
#pragma unroll
  for (int i = 0; i < 3; i++) {
    float xv = x[(size_t)row * DMODEL + t + i * 256];
    vals[i] = xv; v0 += xv; v1 += xv * xv;
  }
  s1[t] = v0; s2[t] = v1;
  __syncthreads();
  for (int s = 128; s > 0; s >>= 1) {
    if (t < s) { s1[t] += s1[t + s]; s2[t] += s2[t + s]; }
    __syncthreads();
  }
  const float mu   = s1[0] * (1.f / DMODEL);
  const float var  = s2[0] * (1.f / DMODEL) - mu * mu;
  const float rstd = rsqrtf(var + 1e-5f);
#pragma unroll
  for (int i = 0; i < 3; i++) {
    int c = t + i * 256;
    out[(size_t)row * DMODEL + c] = f2bf((vals[i] - mu) * rstd * g[c] + b[c]);
  }
}

// ---------------------------------------------------------------------------
// Tiled WMMA GEMM:  C[M,Nc] = A[M,K](bf16,row-major) * B[Nc,K]^T(bf16) + bias
// mode 0: C = acc+bias                    (QKV projection)
// mode 1: C = res + acc+bias              (attention out-proj + residual)
// mode 2: C = res + quickgelu(acc+bias)   (MLP fc + residual)
// Block tile 64x128, 256 threads = 8 waves, each wave -> 16x64 of C
// (one A fragment shared across 4 WMMAs -> 4 wmma per 10 ds_load_b128).
// ---------------------------------------------------------------------------
__global__ __launch_bounds__(256) void gemm_kernel(const __bf16* __restrict__ A,
                                                   const __bf16* __restrict__ B,
                                                   const float*  __restrict__ bias,
                                                   const float*  __restrict__ res,
                                                   float* __restrict__ C,
                                                   int Nc, int K, int mode) {
  __shared__ __bf16 As[64][48];   // pad 48 keeps every [r][8k] slot 16B aligned
  __shared__ __bf16 Bs[128][48];
  const int t    = threadIdx.x;
  const int wave = t >> 5, lane = t & 31, hl = lane >> 4, r = lane & 15;
  const int wm   = wave & 3, wn = wave >> 2;        // 4 x 2 wave grid
  const int bm   = blockIdx.y, bn = blockIdx.x;

  v8f acc[4] = {};

  const int    ar   = t >> 2;          // 0..63  A tile row
  const int    acg  = (t & 3) << 3;    // 8-bf16 (16B) column group
  const size_t arow = (size_t)(bm * 64 + ar) * K;
  const int    br0  = t >> 2;          // B rows br0 and br0+64
  const size_t brow0 = (size_t)(bn * 128 + br0) * K;
  const size_t brow1 = (size_t)(bn * 128 + br0 + 64) * K;

  for (int k0 = 0; k0 < K; k0 += 32) {
    __syncthreads();                    // everyone done reading previous tiles
#if HAVE_ASYNC_LDS
    async_b128(&A[arow + k0 + acg],  &As[ar][acg]);
    async_b128(&B[brow0 + k0 + acg], &Bs[br0][acg]);
    async_b128(&B[brow1 + k0 + acg], &Bs[br0 + 64][acg]);
    wait_async0();                      // this wave's ASYNCcnt -> 0
#else
    *(v8bf*)&As[ar][acg]       = *(const v8bf*)&A[arow + k0 + acg];
    *(v8bf*)&Bs[br0][acg]      = *(const v8bf*)&B[brow0 + k0 + acg];
    *(v8bf*)&Bs[br0 + 64][acg] = *(const v8bf*)&B[brow1 + k0 + acg];
#endif
    if (k0 + 32 < K) {
      __builtin_prefetch(&A[arow + k0 + 32 + acg], 0, 1);   // global_prefetch_b8
      __builtin_prefetch(&B[brow0 + k0 + 32 + acg], 0, 1);
      __builtin_prefetch(&B[brow1 + k0 + 32 + acg], 0, 1);
    }
    __syncthreads();                    // all waves' tile writes visible

    v16bf a, bfr0, bfr1, bfr2, bfr3;
#pragma unroll
    for (int e = 0; e < 16; e++) {
      const int ki = kmap(e, hl);
      a[e]    = As[wm * 16 + r][ki];
      bfr0[e] = Bs[wn * 64 + r][ki];
      bfr1[e] = Bs[wn * 64 + 16 + r][ki];
      bfr2[e] = Bs[wn * 64 + 32 + r][ki];
      bfr3[e] = Bs[wn * 64 + 48 + r][ki];
    }
    acc[0] = __builtin_amdgcn_wmma_f32_16x16x32_bf16(false, a, false, bfr0,
                                                     (short)0, acc[0], false, false);
    acc[1] = __builtin_amdgcn_wmma_f32_16x16x32_bf16(false, a, false, bfr1,
                                                     (short)0, acc[1], false, false);
    acc[2] = __builtin_amdgcn_wmma_f32_16x16x32_bf16(false, a, false, bfr2,
                                                     (short)0, acc[2], false, false);
    acc[3] = __builtin_amdgcn_wmma_f32_16x16x32_bf16(false, a, false, bfr3,
                                                     (short)0, acc[3], false, false);
  }

  const int row0 = bm * 64 + wm * 16 + hl * 8;   // C layout: M = v + 8*half
  const int col0 = bn * 128 + wn * 64 + r;       // N = lane%16, 4 subtiles
#pragma unroll
  for (int v = 0; v < 8; v++) {
    const size_t o = (size_t)(row0 + v) * Nc + col0;
#pragma unroll
    for (int j = 0; j < 4; j++) {
      float c = acc[j][v] + bias[col0 + j * 16];
      if (mode == 1)
        c += res[o + j * 16];
      else if (mode == 2)
        c = res[o + j * 16] + c * (1.f / (1.f + __expf(-1.702f * c)));
      C[o + j * 16] = c;
    }
  }
}

// ---------------------------------------------------------------------------
// Repack fp32 QKV [L*N, 2304] into per-(n,h) bf16 tensors padded to LPAD:
//   Qb,Kb : [NHPAIR][LPAD][64]   (row-major, zero-padded rows >= 257)
//   Vt    : [NHPAIR][64][LPAD]   (transposed so A*V fits the A*B^T pattern)
// ---------------------------------------------------------------------------
__global__ __launch_bounds__(256) void repack_kernel(const float* __restrict__ qkv,
                                                     __bf16* __restrict__ Qb,
                                                     __bf16* __restrict__ Kb,
                                                     __bf16* __restrict__ Vt) {
  const int idx = blockIdx.x * 256 + threadIdx.x;
  const int total = NHPAIR * LPAD * HDIM;
  if (idx >= total) return;
  const int d = idx & 63;
  const int l = (idx >> 6) % LPAD;
  const int p = idx / (LPAD * HDIM);       // p = n*NHEADS + h
  const int n = p / NHEADS, h = p % NHEADS;

  __bf16 q = __builtin_bit_cast(__bf16, (unsigned short)0);
  __bf16 k = q, v = q;
  if (l < LDIM) {
    const size_t base = ((size_t)l * NBATCH + n) * D3 + h * HDIM + d;
    q = f2bf(qkv[base]);
    k = f2bf(qkv[base + DMODEL]);
    v = f2bf(qkv[base + 2 * DMODEL]);
  }
  const size_t pb = (size_t)p * LPAD * HDIM;
  Qb[pb + (size_t)l * HDIM + d] = q;
  Kb[pb + (size_t)l * HDIM + d] = k;
  Vt[pb + (size_t)d * LPAD + l] = v;
}

// ---------------------------------------------------------------------------
// Fused attention: per (n, h, 16-query strip):
//   S = (Q K^T)/8  -> softmax over 257 keys -> weights to d_out
//   -> attn = softmax * V (WMMA) -> bf16 attn rows [l*N+n, h*64+d]
// ---------------------------------------------------------------------------
__global__ __launch_bounds__(256) void attn_kernel(const __bf16* __restrict__ Qb,
                                                   const __bf16* __restrict__ Kb,
                                                   const __bf16* __restrict__ Vt,
                                                   float* __restrict__ wout,
                                                   __bf16* __restrict__ attnb) {
  const int strip = blockIdx.x;          // 0..16 (16 query rows each)
  const int h     = blockIdx.y;
  const int n     = blockIdx.z;
  const int t     = threadIdx.x;
  const int wave  = t >> 5, lane = t & 31, hl = lane >> 4, r = lane & 15;
  const int row0  = strip * 16;
  const int pair  = n * NHEADS + h;
  const size_t pbase = (size_t)pair * LPAD * HDIM;

  __shared__ __bf16 Qs[16][64];
  __shared__ float  Ss[16][LPAD + 1];
  __shared__ __bf16 Ws[16][LPAD];
  __shared__ float  red[16][16];
  __shared__ float  rmax[16], rsum[16];

  for (int i = t; i < 16 * 64; i += 256)
    Qs[i >> 6][i & 63] = Qb[pbase + (size_t)(row0 + (i >> 6)) * HDIM + (i & 63)];
  __syncthreads();

  // ---- scores: S[16, LPAD] = Q[16,64] * K^T, scaled by 1/sqrt(64) ----
  for (int ct = wave; ct < LPAD / 16; ct += 8) {
    v8f acc = {};
#pragma unroll
    for (int kk = 0; kk < 2; kk++) {
      v16bf a, bfr;
#pragma unroll
      for (int e = 0; e < 16; e++) {
        const int ki = kk * 32 + kmap(e, hl);
        a[e]   = Qs[r][ki];
        bfr[e] = Kb[pbase + (size_t)(ct * 16 + r) * HDIM + ki];
      }
      acc = __builtin_amdgcn_wmma_f32_16x16x32_bf16(false, a, false, bfr,
                                                    (short)0, acc, false, false);
    }
#pragma unroll
    for (int v = 0; v < 8; v++)
      Ss[v + 8 * hl][ct * 16 + r] = acc[v] * 0.125f;
  }
  __syncthreads();

  // ---- softmax over valid 257 keys (16 threads per query row) ----
  const int row = t >> 4, sub = t & 15;
  float pm = -1e30f;
  for (int c = sub; c < LDIM; c += 16) pm = fmaxf(pm, Ss[row][c]);
  red[row][sub] = pm;
  __syncthreads();
  if (sub == 0) {
    float m = red[row][0];
#pragma unroll
    for (int i = 1; i < 16; i++) m = fmaxf(m, red[row][i]);
    rmax[row] = m;
  }
  __syncthreads();
  float ps = 0.f;
  for (int c = sub; c < LDIM; c += 16) ps += __expf(Ss[row][c] - rmax[row]);
  red[row][sub] = ps;
  __syncthreads();
  if (sub == 0) {
    float s = 0.f;
#pragma unroll
    for (int i = 0; i < 16; i++) s += red[row][i];
    rsum[row] = 1.f / s;
  }
  __syncthreads();
  const int    gl    = row0 + row;
  const size_t wbase = ((size_t)pair * LDIM + gl) * LDIM;
  for (int c = sub; c < LPAD; c += 16) {
    float w = 0.f;
    if (c < LDIM) {
      w = __expf(Ss[row][c] - rmax[row]) * rsum[row];
      if (gl < LDIM) wout[wbase + c] = w;   // attention-weights output
    }
    Ws[row][c] = f2bf(w);
  }
  __syncthreads();

  // ---- attn strip [16,64] = Ws[16,LPAD] * V   (B = Vt so it is A*B^T) ----
  if (wave < 4) {
    const int ct = wave;                    // 16-wide head-dim tile
    v8f acc = {};
#pragma unroll
    for (int kk = 0; kk < LPAD / 32; kk++) {
      v16bf a, bfr;
#pragma unroll
      for (int e = 0; e < 16; e++) {
        const int ki = kk * 32 + kmap(e, hl);
        a[e]   = Ws[r][ki];
        bfr[e] = Vt[pbase + (size_t)(ct * 16 + r) * LPAD + ki];
      }
      acc = __builtin_amdgcn_wmma_f32_16x16x32_bf16(false, a, false, bfr,
                                                    (short)0, acc, false, false);
    }
#pragma unroll
    for (int v = 0; v < 8; v++) {
      const int ol = row0 + v + 8 * hl;
      if (ol < LDIM)
        attnb[((size_t)ol * NBATCH + n) * DMODEL + h * HDIM + ct * 16 + r] =
            f2bf(acc[v]);
    }
  }
}

// ---------------------------------------------------------------------------
extern "C" void kernel_launch(void* const* d_in, const int* in_sizes, int n_in,
                              void* d_out, int out_size, void* d_ws, size_t ws_size,
                              hipStream_t stream) {
  (void)in_sizes; (void)n_in; (void)out_size; (void)ws_size;
  const float* x     = (const float*)d_in[0];
  const float* in_w  = (const float*)d_in[1];
  const float* in_b  = (const float*)d_in[2];
  const float* ow    = (const float*)d_in[3];
  const float* ob    = (const float*)d_in[4];
  const float* ln1g  = (const float*)d_in[5];
  const float* ln1b  = (const float*)d_in[6];
  const float* ln2g  = (const float*)d_in[7];
  const float* ln2b  = (const float*)d_in[8];
  const float* fcw   = (const float*)d_in[9];
  const float* fcb   = (const float*)d_in[10];

  float* xout = (float*)d_out;                          // [L,N,D]
  float* wout = xout + (size_t)MROWS * DMODEL;          // [N,H,L,L]

  char*  p   = (char*)d_ws;
  size_t off = 0;
  auto take = [&](size_t bytes) -> char* {
    char* r = p + off;
    off = (off + bytes + 255) & ~(size_t)255;
    return r;
  };
  __bf16* Wb_in  = (__bf16*)take((size_t)D3 * DMODEL * 2);
  __bf16* Wb_out = (__bf16*)take((size_t)DMODEL * DMODEL * 2);
  __bf16* Wb_fc  = (__bf16*)take((size_t)DMODEL * DMODEL * 2);
  __bf16* hb     = (__bf16*)take((size_t)MROWS * DMODEL * 2);   // reused 3x
  float*  qkv    = (float*)take((size_t)MROWS * D3 * 4);        // reused
  __bf16* Qb     = (__bf16*)take((size_t)NHPAIR * LPAD * HDIM * 2);
  __bf16* Kb     = (__bf16*)take((size_t)NHPAIR * LPAD * HDIM * 2);
  __bf16* Vt     = (__bf16*)take((size_t)NHPAIR * LPAD * HDIM * 2);

  float*  xattn = qkv;   // qkv fp32 dead after repack -> holds x + attn_out
  __bf16* attnb = hb;    // h(ln1) dead after QKV GEMM -> holds bf16 attn
  __bf16* h2b   = hb;    // attn dead after out-proj GEMM -> holds bf16 ln2(x)

  cvt_bf16_kernel<<<(D3 * DMODEL + 255) / 256, 256, 0, stream>>>(in_w, Wb_in, D3 * DMODEL);
  cvt_bf16_kernel<<<(DMODEL * DMODEL + 255) / 256, 256, 0, stream>>>(ow, Wb_out, DMODEL * DMODEL);
  cvt_bf16_kernel<<<(DMODEL * DMODEL + 255) / 256, 256, 0, stream>>>(fcw, Wb_fc, DMODEL * DMODEL);

  ln_kernel<<<MROWS, 256, 0, stream>>>(x, ln1g, ln1b, hb);

  gemm_kernel<<<dim3(D3 / 128, MROWS / 64), 256, 0, stream>>>(
      hb, Wb_in, in_b, nullptr, qkv, D3, DMODEL, 0);

  repack_kernel<<<(NHPAIR * LPAD * HDIM) / 256, 256, 0, stream>>>(qkv, Qb, Kb, Vt);

  attn_kernel<<<dim3(17, NHEADS, NBATCH), 256, 0, stream>>>(Qb, Kb, Vt, wout, attnb);

  gemm_kernel<<<dim3(DMODEL / 128, MROWS / 64), 256, 0, stream>>>(
      attnb, Wb_out, ob, x, xattn, DMODEL, DMODEL, 1);

  ln_kernel<<<MROWS, 256, 0, stream>>>(xattn, ln2g, ln2b, h2b);

  gemm_kernel<<<dim3(DMODEL / 128, MROWS / 64), 256, 0, stream>>>(
      h2b, Wb_fc, fcb, xattn, xout, DMODEL, DMODEL, 2);
}
